// Attention_16509854286348
// MI455X (gfx1250) — matrix operand order repaired and checked by reading, not compiled, not run on hardware
//
#include <hip/hip_runtime.h>
#include <hip/hip_bf16.h>

// ---------------------------------------------------------------------------
// Attention block for MI455X (gfx1250, wave32, WMMA + async-LDS + TDM).
// fp32 inputs -> bf16 WMMA (16x16x32) with fp32 accumulation.
// ---------------------------------------------------------------------------

typedef __bf16 bf16_t;
typedef bf16_t v16bf __attribute__((ext_vector_type(16)));
typedef float  v8f   __attribute__((ext_vector_type(8)));
typedef unsigned int v4u __attribute__((ext_vector_type(4)));
typedef unsigned int u32x4 __attribute__((ext_vector_type(4)));
typedef int i32x4 __attribute__((ext_vector_type(4)));
typedef int i32x8 __attribute__((ext_vector_type(8)));
typedef int v4i_t __attribute__((ext_vector_type(4)));
typedef unsigned short ushort_t;

typedef __attribute__((address_space(1))) v4i_t gv4i;     // global int4
typedef __attribute__((address_space(3))) v4i_t lv4i;     // LDS int4
typedef __attribute__((address_space(3))) ushort_t lds_ushort;

#define B_   2
#define S_   2048
#define DM_  2048
#define H_   16
#define DH_  128
#define EPS_ 1e-6f

#if defined(__gfx1250__) && __has_builtin(__builtin_amdgcn_global_load_async_to_lds_b128)
#define HAS_ASYNC_LDS 1
#else
#define HAS_ASYNC_LDS 0
#endif

#if defined(__gfx1250__) && __has_builtin(__builtin_amdgcn_tensor_load_to_lds)
#define HAS_TDM 1
#else
#define HAS_TDM 0
#endif

union Frag { v4u q[2]; v16bf f; };        // 32 bytes = 16 bf16 = 8 VGPRs
union Pack8 { v4u v; ushort_t s[8]; };

__device__ __forceinline__ ushort_t f2bf(float f) {
  unsigned u = __float_as_uint(f);
  unsigned r = (u + 0x7FFFu + ((u >> 16) & 1u)) >> 16;   // RNE
  return (ushort_t)r;
}

// 16B global -> LDS, no VGPR staging (ASYNCcnt) when available.
__device__ __forceinline__ void cp_async_b128(ushort_t* lds_dst, const ushort_t* gsrc) {
#if HAS_ASYNC_LDS
  __builtin_amdgcn_global_load_async_to_lds_b128((gv4i*)gsrc, (lv4i*)lds_dst, 0, 0);
#else
  *(v4u*)lds_dst = *(const v4u*)gsrc;
#endif
}

__device__ __forceinline__ void cp_async_wait() {
#if HAS_ASYNC_LDS
#if __has_builtin(__builtin_amdgcn_s_wait_asynccnt)
  __builtin_amdgcn_s_wait_asynccnt(0);
#else
  asm volatile("s_wait_asynccnt 0x0" ::: "memory");
#endif
#endif
}

// TDM: DMA a 2D tile (rows x cols bf16, row stride in elements) into LDS.
// D# layout per CDNA5 ISA ch.8 (group0: count/lds/global/type, group1: dims).
// 6-arg builtin form: (g0, g1, g2, g3, g4, cpol).
__device__ __forceinline__ void tdm_load_2d(ushort_t* lds_dst, const ushort_t* gsrc,
                                            int rows, int cols, int row_stride_elems) {
#if HAS_TDM
  unsigned lds_off = (unsigned)(unsigned long long)(lds_ushort*)lds_dst;
  unsigned long long ga = (unsigned long long)gsrc;
  u32x4 g0 = { 1u,                                          // count=1, no gather
               lds_off,                                     // lds_addr
               (unsigned)(ga & 0xffffffffu),                // global_addr[31:0]
               (unsigned)((ga >> 32) & 0x01ffffffu) | (2u << 30) };  // addr[56:32] | type=2
  i32x8 g1 = { (int)0x00010000,                             // data_size=1 (2 bytes)
               (int)((cols & 0xffff) << 16),                // tensor_dim0[15:0]
               (int)(((unsigned)cols >> 16) | (((unsigned)rows & 0xffffu) << 16)),
               (int)((((unsigned)rows >> 16) & 0xffffu) | (((unsigned)cols & 0xffffu) << 16)),
               (int)(rows & 0xffff),                        // tile_dim1
               (int)row_stride_elems,                       // tensor_dim0_stride[31:0]
               0, 0 };
  i32x4 z4 = { 0, 0, 0, 0 };
  i32x8 z8 = { 0, 0, 0, 0, 0, 0, 0, 0 };
  __builtin_amdgcn_tensor_load_to_lds(g0, g1, z4, z4, z8, 0);
#else
  (void)lds_dst; (void)gsrc; (void)rows; (void)cols; (void)row_stride_elems;
#endif
}

__device__ __forceinline__ void tdm_wait() {
#if HAS_TDM
#if __has_builtin(__builtin_amdgcn_s_wait_tensorcnt)
  __builtin_amdgcn_s_wait_tensorcnt((short)0);
#else
  asm volatile("s_wait_tensorcnt 0x0" ::: "memory");
#endif
#endif
}

// ---------------------------------------------------------------------------
// fp32 -> bf16 conversion (vectorized x4)
// ---------------------------------------------------------------------------
__global__ __launch_bounds__(256) void cvt_f32_bf16(
    const float* __restrict__ src, ushort_t* __restrict__ dst, int n) {
  int i = (blockIdx.x * 256 + threadIdx.x) * 4;
  if (i + 3 < n) {
    float4 v = *(const float4*)(src + i);
    uint2 o;
    o.x = (unsigned)f2bf(v.x) | ((unsigned)f2bf(v.y) << 16);
    o.y = (unsigned)f2bf(v.z) | ((unsigned)f2bf(v.w) << 16);
    *(uint2*)(dst + i) = o;
  }
}

// ---------------------------------------------------------------------------
// bf16 GEMM:  C[M,N] = A[M,K] * B[N,K]^T   (both operands row-major over K)
// block tile 128x64, 8 waves (4x2), wave tile 32x32, K-step 32.
// Tiles staged with GLOBAL_LOAD_ASYNC_TO_LDS (ASYNCcnt).
// ---------------------------------------------------------------------------
#define BM 128
#define BN 64
#define BK 32

__global__ __launch_bounds__(256) void gemm_bf16(
    const ushort_t* __restrict__ A, const ushort_t* __restrict__ Bm,
    float* __restrict__ C, int M, int N, int K) {
  __shared__ ushort_t Al[BM * BK];   // [m][k]
  __shared__ ushort_t Bl[BN * BK];   // [n][k]

  int tid  = threadIdx.x;
  int wave = tid >> 5;
  int lane = tid & 31;
  int bm = blockIdx.y * BM;
  int bn = blockIdx.x * BN;
  int wm = (wave >> 1) * 32;
  int wn = (wave & 1) * 32;

  v8f acc[2][2];
#pragma unroll
  for (int i = 0; i < 2; i++)
#pragma unroll
    for (int j = 0; j < 2; j++)
#pragma unroll
      for (int e = 0; e < 8; e++) acc[i][j][e] = 0.0f;

  const int m16  = lane & 15;
  const int kb8  = (lane >> 4) * 8;    // A frag: K chunks at kb8, kb8+16
  const int kb16 = (lane >> 4) * 16;   // unused; B frag also reads K chunks at kb8, kb8+16

  const int ar = tid >> 1, ac = (tid & 1) * 16;   // A tile loader coords
  const int br = tid >> 2, bc = (tid & 3) * 8;    // B tile loader coords

  for (int k0 = 0; k0 < K; k0 += BK) {
    __syncthreads();
    // stage A tile: 128 rows x 32 k (each thread 32B) + B tile (16B) async->LDS
    const ushort_t* ga = A + (size_t)(bm + ar) * K + k0 + ac;
    cp_async_b128(&Al[ar * BK + ac], ga);
    cp_async_b128(&Al[ar * BK + ac + 8], ga + 8);
    cp_async_b128(&Bl[br * BK + bc], Bm + (size_t)(bn + br) * K + k0 + bc);
    cp_async_wait();
    __syncthreads();

    if (k0 + BK < K) {  // pull next tiles toward L2 (global_prefetch_b8)
      __builtin_prefetch(A + (size_t)(bm + ar) * K + k0 + BK + ac, 0, 0);
      __builtin_prefetch(Bm + (size_t)(bn + br) * K + k0 + BK + bc, 0, 0);
    }

    Frag af[2], bfr[2];
#pragma unroll
    for (int i = 0; i < 2; i++) {
      int row = wm + i * 16 + m16;
      af[i].q[0] = *(const v4u*)&Al[row * BK + kb8];
      af[i].q[1] = *(const v4u*)&Al[row * BK + kb8 + 16];
    }
#pragma unroll
    for (int j = 0; j < 2; j++) {
      int row = wn + j * 16 + m16;
      bfr[j].q[0] = *(const v4u*)&Bl[row * BK + kb8];
      bfr[j].q[1] = *(const v4u*)&Bl[row * BK + kb8 + 16];
    }
#pragma unroll
    for (int i = 0; i < 2; i++)
#pragma unroll
      for (int j = 0; j < 2; j++)
        acc[i][j] = __builtin_amdgcn_wmma_f32_16x16x32_bf16(
            false, af[i].f, false, bfr[j].f, (short)0, acc[i][j], false, false);
  }

  // C layout: VGPR v -> row v (lanes 0-15) / row v+8 (lanes 16-31), col = lane&15
  int cn = lane & 15;
  int ro = (lane >> 4) * 8;
#pragma unroll
  for (int i = 0; i < 2; i++)
#pragma unroll
    for (int j = 0; j < 2; j++)
#pragma unroll
      for (int v = 0; v < 8; v++) {
        int gm = bm + wm + i * 16 + ro + v;
        int gn = bn + wn + j * 16 + cn;
        C[(size_t)gm * N + gn] = acc[i][j][v];
      }
}

// ---------------------------------------------------------------------------
// RMSNorm (over DH) + RoPE, fp32 in -> bf16 out. One wave per (b,s,h) row.
// ---------------------------------------------------------------------------
__global__ __launch_bounds__(256) void rmsnorm_rope(
    const float* __restrict__ t, const float* __restrict__ normw,
    const float* __restrict__ rope, ushort_t* __restrict__ outb) {
  int row  = blockIdx.x * 8 + (threadIdx.x >> 5);   // (b*S + s)*H + h
  int lane = threadIdx.x & 31;
  int s = (row / H_) % S_;

  const float* p = t + (size_t)row * DH_;
  int d0 = lane * 4;
  float4 v = *(const float4*)(p + d0);
  float ss = v.x * v.x + v.y * v.y + v.z * v.z + v.w * v.w;
#pragma unroll
  for (int m = 16; m >= 1; m >>= 1) ss += __shfl_xor(ss, m, 32);
  float inv = rsqrtf(ss * (1.0f / DH_) + EPS_);
  float4 w = *(const float4*)(normw + d0);

  float xs[4] = {v.x * inv * w.x, v.y * inv * w.y, v.z * inv * w.z, v.w * inv * w.w};
  float ys[4];
#pragma unroll
  for (int i = 0; i < 4; i++) ys[i] = __shfl_xor(xs[i], 16, 32);  // partner half

  const float* fr = rope + (size_t)s * 256;  // (64 pairs) * 2 * 2
  int j = lane >> 4;                          // 0: d<64 half, 1: d>=64 half
  int pbase = d0 & 63;
  float o[4];
#pragma unroll
  for (int i = 0; i < 4; i++) {
    const float* f = fr + (pbase + i) * 4 + j * 2;
    float x1 = (j == 0) ? xs[i] : ys[i];  // value from first half
    float x2 = (j == 0) ? ys[i] : xs[i];  // value from second half
    o[i] = f[0] * x1 + f[1] * x2;
  }
  uint2 ob;
  ob.x = (unsigned)f2bf(o[0]) | ((unsigned)f2bf(o[1]) << 16);
  ob.y = (unsigned)f2bf(o[2]) | ((unsigned)f2bf(o[3]) << 16);
  *(uint2*)(outb + (size_t)row * DH_ + d0) = ob;
}

// ---------------------------------------------------------------------------
// Flash attention: one block per (b, h, 128 q rows); 8 waves x 16 rows.
// K chunk staged by the Tensor Data Mover (TENSORcnt) when available, else
// async-LDS copies; V staged transposed for contiguous B-fragments.
// ---------------------------------------------------------------------------
#define QB 128
#define KB 64

__global__ __launch_bounds__(256) void attn_kernel(
    const ushort_t* __restrict__ qg, const ushort_t* __restrict__ kg,
    const ushort_t* __restrict__ vg, float* __restrict__ outp) {
  __shared__ ushort_t Kt[KB * DH_];       // [key][d]  16 KB
  __shared__ ushort_t VtT[DH_ * KB];      // [d][key]  16 KB
  __shared__ ushort_t Pw[8][16 * KB];     // per-wave P tile, 16 KB

  int tid = threadIdx.x, wave = tid >> 5, lane = tid & 31;
  int blk  = blockIdx.x;
  int qblk = blk % (S_ / QB);
  int h    = (blk / (S_ / QB)) % H_;
  int b    = blk / ((S_ / QB) * H_);
  const float scale = 0.088388347762f;    // 1/sqrt(128)

  const int m16 = lane & 15, hi = lane >> 4;
  const int kb8 = hi * 8, kb16 = hi * 16;

  // Q fragments for this wave's 16 rows (resident for the whole pass)
  Frag qf[4];
  {
    int qrow = qblk * QB + wave * 16 + m16;
    const ushort_t* qptr = qg + ((size_t)(b * S_ + qrow) * H_ + h) * DH_;
#pragma unroll
    for (int kk = 0; kk < 4; kk++) {
      qf[kk].q[0] = *(const v4u*)(qptr + kk * 32 + kb8);
      qf[kk].q[1] = *(const v4u*)(qptr + kk * 32 + kb8 + 16);
    }
  }

  v8f oacc[8];
#pragma unroll
  for (int n = 0; n < 8; n++)
#pragma unroll
    for (int e = 0; e < 8; e++) oacc[n][e] = 0.0f;
  float rmax[8], rsum[8];
#pragma unroll
  for (int v = 0; v < 8; v++) { rmax[v] = -1e30f; rsum[v] = 0.0f; }

  const int lr = tid >> 2, lc = (tid & 3) * 32;   // chunk loader coords

  for (int s0 = 0; s0 < S_; s0 += KB) {
    __syncthreads();
    {
      // stage V transposed [d][key] (manual transform through VGPRs)
      const ushort_t* gv = vg + ((size_t)(b * S_ + s0 + lr) * H_ + h) * DH_ + lc;
#pragma unroll
      for (int u = 0; u < 4; u++) {
        Pack8 pk; pk.v = *(const v4u*)(gv + u * 8);
#pragma unroll
        for (int e = 0; e < 8; e++) VtT[(lc + u * 8 + e) * KB + lr] = pk.s[e];
      }
      // stage K [key][d]: TDM descriptor from wave 0 (uniform addresses),
      // else per-thread async copies to LDS.
#if HAS_TDM
      if (wave == 0) {
        tdm_load_2d(Kt, kg + ((size_t)(b * S_ + s0) * H_ + h) * DH_,
                    KB, DH_, H_ * DH_);
        tdm_wait();
      }
#else
      const ushort_t* gk = kg + ((size_t)(b * S_ + s0 + lr) * H_ + h) * DH_ + lc;
#pragma unroll
      for (int u = 0; u < 4; u++)
        cp_async_b128(&Kt[lr * DH_ + lc + u * 8], gk + u * 8);
      cp_async_wait();
#endif
    }
    __syncthreads();

    // scores: S = Q(16x128) * K^T(128x64) -> 4 tiles x 4 k-steps = 16 WMMA
    v8f sacc[4];
#pragma unroll
    for (int j = 0; j < 4; j++)
#pragma unroll
      for (int e = 0; e < 8; e++) sacc[j][e] = 0.0f;
#pragma unroll
    for (int j = 0; j < 4; j++) {
      int krow = j * 16 + m16;
#pragma unroll
      for (int kk = 0; kk < 4; kk++) {
        Frag bf_;
        bf_.q[0] = *(const v4u*)&Kt[krow * DH_ + kk * 32 + kb8];
        bf_.q[1] = *(const v4u*)&Kt[krow * DH_ + kk * 32 + kb8 + 16];
        sacc[j] = __builtin_amdgcn_wmma_f32_16x16x32_bf16(
            false, qf[kk].f, false, bf_.f, (short)0, sacc[j], false, false);
      }
    }

    // online softmax per row (rows live in the 16-lane column groups)
#pragma unroll
    for (int v = 0; v < 8; v++) {
      float mloc = -1e30f;
#pragma unroll
      for (int j = 0; j < 4; j++) mloc = fmaxf(mloc, sacc[j][v] * scale);
#pragma unroll
      for (int m = 1; m < 16; m <<= 1) mloc = fmaxf(mloc, __shfl_xor(mloc, m, 32));
      float mnew  = fmaxf(rmax[v], mloc);
      float alpha = __expf(rmax[v] - mnew);
      rmax[v] = mnew;
      float psum = 0.0f;
#pragma unroll
      for (int j = 0; j < 4; j++) {
        float pv = __expf(sacc[j][v] * scale - mnew);
        sacc[j][v] = pv;
        psum += pv;
      }
#pragma unroll
      for (int m = 1; m < 16; m <<= 1) psum += __shfl_xor(psum, m, 32);
      rsum[v] = rsum[v] * alpha + psum;
#pragma unroll
      for (int n = 0; n < 8; n++) oacc[n][v] *= alpha;
    }

    // repack P (16x64) into A-fragment layout through per-wave LDS scratch
    ushort_t* P = &Pw[wave][0];
#pragma unroll
    for (int j = 0; j < 4; j++)
#pragma unroll
      for (int v = 0; v < 8; v++)
        P[(v + hi * 8) * KB + (j * 16 + m16)] = f2bf(sacc[j][v]);

    // O += P(16x64) * V(64x128): 2 k-steps x 8 d-tiles = 16 WMMA
#pragma unroll
    for (int kk = 0; kk < 2; kk++) {
      Frag pf;
      pf.q[0] = *(const v4u*)&P[m16 * KB + kk * 32 + kb8];
      pf.q[1] = *(const v4u*)&P[m16 * KB + kk * 32 + kb8 + 16];
#pragma unroll
      for (int n = 0; n < 8; n++) {
        Frag vf_;
        int drow = n * 16 + m16;
        vf_.q[0] = *(const v4u*)&VtT[drow * KB + kk * 32 + kb8];
        vf_.q[1] = *(const v4u*)&VtT[drow * KB + kk * 32 + kb8 + 16];
        oacc[n] = __builtin_amdgcn_wmma_f32_16x16x32_bf16(
            false, pf.f, false, vf_.f, (short)0, oacc[n], false, false);
      }
    }
  }

  // epilogue: normalize by row sum, scatter to (B,S,H*DH)
#pragma unroll
  for (int n = 0; n < 8; n++)
#pragma unroll
    for (int v = 0; v < 8; v++) {
      int gq = qblk * QB + wave * 16 + hi * 8 + v;
      int gd = h * DH_ + n * 16 + m16;
      outp[(size_t)(b * S_ + gq) * (H_ * DH_) + gd] = oacc[n][v] / rsum[v];
    }
}

// ---------------------------------------------------------------------------
// Launch
// ---------------------------------------------------------------------------
extern "C" void kernel_launch(void* const* d_in, const int* in_sizes, int n_in,
                              void* d_out, int out_size, void* d_ws, size_t ws_size,
                              hipStream_t stream) {
  (void)in_sizes; (void)n_in; (void)out_size; (void)ws_size;
  const float* x    = (const float*)d_in[0];
  const float* rope = (const float*)d_in[1];
  const float* wq   = (const float*)d_in[2];
  const float* wk   = (const float*)d_in[3];
  const float* wv   = (const float*)d_in[4];
  const float* wo   = (const float*)d_in[5];
  const float* qnw  = (const float*)d_in[6];
  const float* knw  = (const float*)d_in[7];
  float* out = (float*)d_out;

  char* ws = (char*)d_ws;
  size_t off = 0;
  auto nxt = [&](size_t bytes) -> char* {
    char* p = ws + off;
    off += (bytes + 255) & ~(size_t)255;
    return p;
  };
  const size_t nAct = (size_t)B_ * S_ * DM_;   // 8.39 M elements
  const size_t nW   = (size_t)DM_ * DM_;

  ushort_t* xb  = (ushort_t*)nxt(nAct * 2);
  ushort_t* wqb = (ushort_t*)nxt(nW * 2);
  ushort_t* wkb = (ushort_t*)nxt(nW * 2);
  ushort_t* wvb = (ushort_t*)nxt(nW * 2);
  ushort_t* wob = (ushort_t*)nxt(nW * 2);
  float*    qf  = (float*)nxt(nAct * 4);
  float*    kf  = (float*)nxt(nAct * 4);
  float*    vf  = (float*)nxt(nAct * 4);
  ushort_t* qbb = (ushort_t*)nxt(nAct * 2);
  ushort_t* kbb = (ushort_t*)nxt(nAct * 2);
  ushort_t* vbb = (ushort_t*)nxt(nAct * 2);
  float*    attnf = qf;               // reuse: Q fp32 dead after norm/rope
  ushort_t* attnb = (ushort_t*)kf;    // reuse: K fp32 dead after norm/rope

  const int cvtThreads = 256;
  auto cvtBlocks = [](size_t n) { return (unsigned)((n / 4 + 255) / 256); };

  cvt_f32_bf16<<<cvtBlocks(nAct), cvtThreads, 0, stream>>>(x, xb, (int)nAct);
  cvt_f32_bf16<<<cvtBlocks(nW), cvtThreads, 0, stream>>>(wq, wqb, (int)nW);
  cvt_f32_bf16<<<cvtBlocks(nW), cvtThreads, 0, stream>>>(wk, wkb, (int)nW);
  cvt_f32_bf16<<<cvtBlocks(nW), cvtThreads, 0, stream>>>(wv, wvb, (int)nW);
  cvt_f32_bf16<<<cvtBlocks(nW), cvtThreads, 0, stream>>>(wo, wob, (int)nW);

  dim3 ggrid(DM_ / BN, (B_ * S_) / BM);
  gemm_bf16<<<ggrid, 256, 0, stream>>>(xb, wqb, qf, B_ * S_, DM_, DM_);
  gemm_bf16<<<ggrid, 256, 0, stream>>>(xb, wkb, kf, B_ * S_, DM_, DM_);
  gemm_bf16<<<ggrid, 256, 0, stream>>>(xb, wvb, vf, B_ * S_, DM_, DM_);

  rmsnorm_rope<<<B_ * S_ * H_ / 8, 256, 0, stream>>>(qf, qnw, rope, qbb);
  rmsnorm_rope<<<B_ * S_ * H_ / 8, 256, 0, stream>>>(kf, knw, rope, kbb);
  cvt_f32_bf16<<<cvtBlocks(nAct), cvtThreads, 0, stream>>>(vf, vbb, (int)nAct);

  attn_kernel<<<B_ * H_ * (S_ / QB), 256, 0, stream>>>(qbb, kbb, vbb, attnf);

  cvt_f32_bf16<<<cvtBlocks(nAct), cvtThreads, 0, stream>>>(attnf, attnb, (int)nAct);
  gemm_bf16<<<ggrid, 256, 0, stream>>>(attnb, wob, out, B_ * S_, DM_, DM_);
}
